// decoder_model_24670292148674
// MI455X (gfx1250) — compile-verified, matrix operand order checked by reading, not run on hardware
//
#include <hip/hip_runtime.h>

// Problem constants (match reference setup_inputs).
#define B_SZ   256
#define H_SZ   1024
#define I_SZ   1024
#define G3     (3 * H_SZ)
#define NSTEPS 128

// GEMM tiling.
#define BM 64
#define BN 128
#define BK 32
#define LDS_STRIDE 40   // BK + 8 halves padding (80B rows, bank-spread)

typedef __attribute__((ext_vector_type(16))) _Float16 v16h;
typedef __attribute__((ext_vector_type(8)))  _Float16 v8h;
typedef __attribute__((ext_vector_type(8)))  float    v8f;

// Pointer types for the async-to-LDS builtin:
// (v4i addrspace(1)*, v4i addrspace(3)*, imm offset, imm cpol)
typedef int v4i __attribute__((vector_size(16)));
typedef __attribute__((address_space(1))) v4i* gas_v4i_ptr;
typedef __attribute__((address_space(3))) v4i* las_v4i_ptr;

union V16 { v16h v; v8h h[2]; };

// ---------------------------------------------------------------- utilities
__global__ void f32_to_f16_kernel(const float* __restrict__ src,
                                  _Float16* __restrict__ dst, int n) {
  int i = blockIdx.x * blockDim.x + threadIdx.x;
  int stride = gridDim.x * blockDim.x;
  for (; i < n; i += stride) dst[i] = (_Float16)src[i];
}

__global__ void zero_f32_kernel(float* __restrict__ dst, int n) {
  int i = blockIdx.x * blockDim.x + threadIdx.x;
  int stride = gridDim.x * blockDim.x;
  for (; i < n; i += stride) dst[i] = 0.0f;
}

__global__ void zero_f16_kernel(_Float16* __restrict__ dst, int n) {
  int i = blockIdx.x * blockDim.x + threadIdx.x;
  int stride = gridDim.x * blockDim.x;
  for (; i < n; i += stride) dst[i] = (_Float16)0.0f;
}

__global__ void fill_out_kernel(float* __restrict__ out,
                                const float* __restrict__ b_out, int n) {
  int i = blockIdx.x * blockDim.x + threadIdx.x;
  int stride = gridDim.x * blockDim.x;
  float v = b_out[0];
  for (; i < n; i += stride) out[i] = v;
}

// ---------------------------------------------------------------- WMMA GEMM
// C = A (MxK, row-major f16) * B^T where B is (NxK, row-major f16).
// C[m,n] = sum_k A[m,k] * B[n,k].  C is fp32 row-major MxN.
//
// 4 waves (128 threads), block tile 64x128, wave tile 32x64 (2x4 WMMA accums).
// K-slabs of 32 are cooperatively staged into double-buffered LDS with the
// CDNA5 async-to-LDS path (ASYNCcnt + s_wait_asynccnt), fragments are read
// back with ds_load_b128 in the ISA WMMA register layout.
// blockIdx.z selects direction (0: fwd operands, 1: bwd operands).
__global__ __launch_bounds__(128)
void gemm_wmma_kernel(const _Float16* __restrict__ A0, const _Float16* __restrict__ A1,
                      const _Float16* __restrict__ B0w, const _Float16* __restrict__ B1w,
                      float* __restrict__ C0, float* __restrict__ C1,
                      int N, int K) {
  const _Float16* A  = blockIdx.z ? A1  : A0;
  const _Float16* Bw = blockIdx.z ? B1w : B0w;
  float*          C  = blockIdx.z ? C1  : C0;

  __shared__ _Float16 sA[2][BM * LDS_STRIDE];   // 10240 B
  __shared__ _Float16 sB[2][BN * LDS_STRIDE];   // 20480 B

  const int tid  = threadIdx.x;
  const int lane = tid & 31;
  const int wave = tid >> 5;     // 0..3
  const int wm   = wave >> 1;    // 0..1 (M)
  const int wn   = wave & 1;     // 0..1 (N)

  const int m_blk = blockIdx.y * BM;
  const int n_blk = blockIdx.x * BN;

  // Cooperative async copy of one K-slab (BK=32 halves = 64B per row) into
  // LDS buffer `buf`.  Rows are split into 4 x 16B segments; each of the 128
  // threads moves 2 segments of A (64 rows) and 4 segments of B (128 rows).
  auto issue_loads = [&](int buf, int k0) {
#pragma unroll
    for (int i = 0; i < 2; ++i) {
      const int c = tid + i * 128;          // 0..255
      const int row = c >> 2, seg = c & 3;  // 16B segment within the row slab
      __builtin_amdgcn_global_load_async_to_lds_b128(
          (gas_v4i_ptr)(A + (size_t)(m_blk + row) * K + k0 + seg * 8),
          (las_v4i_ptr)&sA[buf][row * LDS_STRIDE + seg * 8],
          0, 0);
    }
#pragma unroll
    for (int i = 0; i < 4; ++i) {
      const int c = tid + i * 128;          // 0..511
      const int row = c >> 2, seg = c & 3;
      __builtin_amdgcn_global_load_async_to_lds_b128(
          (gas_v4i_ptr)(Bw + (size_t)(n_blk + row) * K + k0 + seg * 8),
          (las_v4i_ptr)&sB[buf][row * LDS_STRIDE + seg * 8],
          0, 0);
    }
  };

  v8f acc[2][4] = {};

  // WMMA 16-bit operand layout: lane&15 selects the row (A) / column (B);
  // lane>>4 selects the K half: regs 0..3 hold K = half*8..half*8+7,
  // regs 4..7 hold K = 16+half*8 .. 16+half*8+7.
  const int r   = lane & 15;
  const int hlf = (lane >> 4) * 8;

  issue_loads(0, 0);
  asm volatile("s_wait_asynccnt 0" ::: "memory");
  __syncthreads();

  const int KT = K / BK;
  for (int kt = 0; kt < KT; ++kt) {
    const int cur = kt & 1;
    if (kt + 1 < KT) issue_loads(cur ^ 1, (kt + 1) * BK);

    V16 a[2], b[4];
#pragma unroll
    for (int i = 0; i < 2; ++i) {
      const _Float16* p = &sA[cur][(wm * 32 + i * 16 + r) * LDS_STRIDE + hlf];
      a[i].h[0] = *(const v8h*)p;
      a[i].h[1] = *(const v8h*)(p + 16);
    }
#pragma unroll
    for (int j = 0; j < 4; ++j) {
      const _Float16* p = &sB[cur][(wn * 64 + j * 16 + r) * LDS_STRIDE + hlf];
      b[j].h[0] = *(const v8h*)p;
      b[j].h[1] = *(const v8h*)(p + 16);
    }

#pragma unroll
    for (int i = 0; i < 2; ++i)
#pragma unroll
      for (int j = 0; j < 4; ++j)
        acc[i][j] = __builtin_amdgcn_wmma_f32_16x16x32_f16(
            false, a[i].v, false, b[j].v, (short)0, acc[i][j], false, false);

    // Own-wave async loads done, then block-wide barrier: next slab resident,
    // current slab fully consumed before it is overwritten in the next iter.
    asm volatile("s_wait_asynccnt 0" ::: "memory");
    __syncthreads();
  }

  // C/D 16x16 f32 layout: reg v, lane l -> row = v + (l>>4)*8, col = l&15.
  const int row_off = (lane >> 4) * 8;
  const int col     = lane & 15;
#pragma unroll
  for (int i = 0; i < 2; ++i) {
    const int mrow = m_blk + wm * 32 + i * 16 + row_off;
#pragma unroll
    for (int j = 0; j < 4; ++j) {
      const int ncol = n_blk + wn * 64 + j * 16 + col;
#pragma unroll
      for (int v = 0; v < 8; ++v)
        C[(size_t)(mrow + v) * N + ncol] = acc[i][j][v];
    }
  }
}

// ---------------------------------------------------------------- GRU gates
// grid: (H/256, B, 2 dirs), block 256.  Updates h in place (fp32 + f16 shadow)
// and accumulates this step's output projection into out[m*NSTEPS + t].
__global__ __launch_bounds__(256)
void gru_gate_kernel(const float* __restrict__ gh_f, const float* __restrict__ gh_b,
                     const float* __restrict__ gi_f, const float* __restrict__ gi_b,
                     const float* __restrict__ bi_f, const float* __restrict__ bh_f,
                     const float* __restrict__ bi_b, const float* __restrict__ bh_b,
                     float* __restrict__ h_f, float* __restrict__ h_b,
                     _Float16* __restrict__ h16_f, _Float16* __restrict__ h16_b,
                     const float* __restrict__ w_out,
                     float* __restrict__ out, int t) {
  const int dir = blockIdx.z;
  const float* gh = dir ? gh_b : gh_f;
  const float* gi = dir ? gi_b : gi_f;
  const float* bi = dir ? bi_b : bi_f;
  const float* bh = dir ? bh_b : bh_f;
  float*       h  = dir ? h_b  : h_f;
  _Float16*  h16  = dir ? h16_b : h16_f;

  const int m = blockIdx.y;
  const int j = blockIdx.x * 256 + threadIdx.x;

  const size_t rowg = (size_t)m * G3;
  const float gr  = gi[rowg + j]            + bi[j]            + gh[rowg + j]            + bh[j];
  const float gz  = gi[rowg + H_SZ + j]     + bi[H_SZ + j]     + gh[rowg + H_SZ + j]     + bh[H_SZ + j];
  const float gin = gi[rowg + 2 * H_SZ + j] + bi[2 * H_SZ + j];
  const float ghn = gh[rowg + 2 * H_SZ + j] + bh[2 * H_SZ + j];

  const float rr = 1.0f / (1.0f + __expf(-gr));
  const float zz = 1.0f / (1.0f + __expf(-gz));
  const float ng = tanhf(gin + rr * ghn);

  const size_t hidx = (size_t)m * H_SZ + j;
  const float ho = h[hidx];
  const float hn = (1.0f - zz) * ng + zz * ho;
  h[hidx]   = hn;
  h16[hidx] = (_Float16)hn;

  // Partial output projection: out[m,t] += w_out[dir*H + j] * h_new.
  float contrib = w_out[dir * H_SZ + j] * hn;
  __shared__ float red[256];
  red[threadIdx.x] = contrib;
  __syncthreads();
  for (int s = 128; s > 0; s >>= 1) {
    if (threadIdx.x < s) red[threadIdx.x] += red[threadIdx.x + s];
    __syncthreads();
  }
  if (threadIdx.x == 0) atomicAdd(&out[(size_t)m * NSTEPS + t], red[0]);
}

// ---------------------------------------------------------------- launch
extern "C" void kernel_launch(void* const* d_in, const int* in_sizes, int n_in,
                              void* d_out, int out_size, void* d_ws, size_t ws_size,
                              hipStream_t stream) {
  (void)in_sizes; (void)n_in; (void)out_size; (void)ws_size;

  const float* x      = (const float*)d_in[0];   // (256,1,1024) == x0
  const float* w_ih_f = (const float*)d_in[2];   // (3072,1024)
  const float* w_hh_f = (const float*)d_in[3];   // (3072,1024)
  const float* b_ih_f = (const float*)d_in[4];   // (3072)
  const float* b_hh_f = (const float*)d_in[5];
  const float* w_ih_b = (const float*)d_in[6];
  const float* w_hh_b = (const float*)d_in[7];
  const float* b_ih_b = (const float*)d_in[8];
  const float* b_hh_b = (const float*)d_in[9];
  const float* w_out  = (const float*)d_in[10];  // (1,2048)
  const float* b_out  = (const float*)d_in[11];  // (1)
  float* out = (float*)d_out;                    // (256,128,1)

  // Workspace carve-out (256B aligned).
  char* ws = (char*)d_ws;
  size_t off = 0;
  auto carve = [&](size_t bytes) -> void* {
    void* p = ws + off;
    off = (off + bytes + 255) & ~(size_t)255;
    return p;
  };
  const size_t wN = (size_t)G3 * H_SZ;          // 3,145,728 elems
  _Float16* whh16_f = (_Float16*)carve(wN * 2);
  _Float16* whh16_b = (_Float16*)carve(wN * 2);
  _Float16* wih16_f = (_Float16*)carve(wN * 2);
  _Float16* wih16_b = (_Float16*)carve(wN * 2);
  _Float16* x16     = (_Float16*)carve((size_t)B_SZ * I_SZ * 2);
  float*    gi_f    = (float*)carve((size_t)B_SZ * G3 * 4);
  float*    gi_b    = (float*)carve((size_t)B_SZ * G3 * 4);
  float*    gh_f    = (float*)carve((size_t)B_SZ * G3 * 4);
  float*    gh_b    = (float*)carve((size_t)B_SZ * G3 * 4);
  float*    h_f     = (float*)carve((size_t)B_SZ * H_SZ * 4);
  float*    h_b     = (float*)carve((size_t)B_SZ * H_SZ * 4);
  _Float16* h16_f   = (_Float16*)carve((size_t)B_SZ * H_SZ * 2);
  _Float16* h16_b   = (_Float16*)carve((size_t)B_SZ * H_SZ * 2);

  // ---- one-time setup ------------------------------------------------
  const int wn_i = (int)wN;                 // 3,145,728
  f32_to_f16_kernel<<<2048, 256, 0, stream>>>(w_hh_f, whh16_f, wn_i);
  f32_to_f16_kernel<<<2048, 256, 0, stream>>>(w_hh_b, whh16_b, wn_i);
  f32_to_f16_kernel<<<2048, 256, 0, stream>>>(w_ih_f, wih16_f, wn_i);
  f32_to_f16_kernel<<<2048, 256, 0, stream>>>(w_ih_b, wih16_b, wn_i);
  f32_to_f16_kernel<<<256, 256, 0, stream>>>(x, x16, B_SZ * I_SZ);

  zero_f32_kernel<<<256, 256, 0, stream>>>(h_f, B_SZ * H_SZ);
  zero_f32_kernel<<<256, 256, 0, stream>>>(h_b, B_SZ * H_SZ);
  zero_f16_kernel<<<256, 256, 0, stream>>>(h16_f, B_SZ * H_SZ);
  zero_f16_kernel<<<256, 256, 0, stream>>>(h16_b, B_SZ * H_SZ);
  fill_out_kernel<<<128, 256, 0, stream>>>(out, b_out, B_SZ * NSTEPS);

  // gi = x0 @ w_ih^T (raw; biases added in the gate kernel each step)
  dim3 ggrid(G3 / BN, B_SZ / BM, 2);
  gemm_wmma_kernel<<<ggrid, 128, 0, stream>>>(x16, x16, wih16_f, wih16_b,
                                              gi_f, gi_b, G3, I_SZ);

  // ---- recurrence ----------------------------------------------------
  dim3 egrid(H_SZ / 256, B_SZ, 2);
  for (int t = 0; t < NSTEPS; ++t) {
    gemm_wmma_kernel<<<ggrid, 128, 0, stream>>>(h16_f, h16_b, whh16_f, whh16_b,
                                                gh_f, gh_b, G3, H_SZ);
    gru_gate_kernel<<<egrid, 256, 0, stream>>>(gh_f, gh_b, gi_f, gi_b,
                                               b_ih_f, b_hh_f, b_ih_b, b_hh_b,
                                               h_f, h_b, h16_f, h16_b,
                                               w_out, out, t);
  }
}
